// InterferenceAttention_21955872817275
// MI455X (gfx1250) — matrix operand order repaired and checked by reading, not compiled, not run on hardware
//
#include <hip/hip_runtime.h>
#include <hip/hip_bf16.h>

typedef __attribute__((ext_vector_type(16))) __bf16 v16bf;
typedef __attribute__((ext_vector_type(8)))  float  v8f;
typedef __attribute__((ext_vector_type(4)))  unsigned int u32x4;
typedef __attribute__((ext_vector_type(8)))  int i32x8;
typedef __attribute__((ext_vector_type(4)))  int i32x4;

__device__ __forceinline__ __bf16 f2bf(float f) {
  union { float f; unsigned int u; } a; a.f = f;
  unsigned int r = a.u + 0x7FFFu + ((a.u >> 16) & 1u);   // round-to-nearest-even
  union { unsigned short s; __bf16 b; } o; o.s = (unsigned short)(r >> 16);
  return o.b;
}

__device__ __forceinline__ v8f vzero8() {
  v8f z = {0.f,0.f,0.f,0.f,0.f,0.f,0.f,0.f}; return z;
}

__device__ __forceinline__ v16bf load_frag2(const __bf16* p0, const __bf16* p1) {
  union { uint4 u[2]; v16bf v; } r;
  r.u[0] = *(const uint4*)p0;
  r.u[1] = *(const uint4*)p1;
  return r.v;
}
__device__ __forceinline__ v16bf load_bfrag(const __bf16* p) { return load_frag2(p, p + 16); }

// ---------------------------------------------------------------------------
// Tensor Data Mover: 2D bf16 tile -> LDS with pad (32DW interval, 4DW amount)
// => 72-element LDS row stride. D# bitfields per CDNA5 ISA 8.3/8.4.
// ---------------------------------------------------------------------------
__device__ __forceinline__ void tdm_load_tile_2d(
    unsigned lds_addr, unsigned long long gaddr,
    unsigned tensor_d0, unsigned tensor_d1,
    unsigned tile_d0, unsigned tile_d1,
    unsigned long long d0_stride) {
  u32x4 g0;
  g0[0] = 1u;                                               // count=1, user mode
  g0[1] = lds_addr;
  g0[2] = (unsigned)(gaddr & 0xFFFFFFFFu);
  g0[3] = (unsigned)((gaddr >> 32) & 0x01FFFFFFu) | (2u << 30); // | type=2

  const unsigned pad_bits = (1u << 20) | (4u << 22) | (3u << 25);
  i32x8 g1;
  g1[0] = (int)((1u << 16) | pad_bits);                     // data_size=2B
  g1[1] = (int)((tensor_d0 & 0xFFFFu) << 16);
  g1[2] = (int)((tensor_d0 >> 16) | ((tensor_d1 & 0xFFFFu) << 16));
  g1[3] = (int)((tensor_d1 >> 16) | (tile_d0 << 16));
  g1[4] = (int)(tile_d1 & 0xFFFFu);
  g1[5] = (int)(d0_stride & 0xFFFFFFFFu);
  g1[6] = (int)((d0_stride >> 32) & 0xFFFFu);
  g1[7] = 0;

  i32x4 z4 = {0, 0, 0, 0};
#if defined(__clang_major__) && (__clang_major__ >= 23)
  i32x8 z8 = {0, 0, 0, 0, 0, 0, 0, 0};
  __builtin_amdgcn_tensor_load_to_lds(g0, g1, z4, z4, z8, 0);
#else
  __builtin_amdgcn_tensor_load_to_lds(g0, g1, z4, z4, 0);
#endif
}

// ---------------------------------------------------------------------------
__global__ void ia_cvt_kernel(const float4* __restrict__ in, uint2* __restrict__ out, int n4) {
  int i = blockIdx.x * blockDim.x + threadIdx.x;
  int stride = gridDim.x * blockDim.x;
  for (; i < n4; i += stride) {
    float4 v = in[i];
    union { __bf16 b[4]; uint2 u; } o;
    o.b[0] = f2bf(v.x); o.b[1] = f2bf(v.y); o.b[2] = f2bf(v.z); o.b[3] = f2bf(v.w);
    out[i] = o.u;
  }
}

// ---------------------------------------------------------------------------
__global__ __launch_bounds__(32) void ia_phase_kernel(
    const float* __restrict__ x, const float* __restrict__ Wp,
    const float* __restrict__ bp, const float* __restrict__ gamma,
    float* __restrict__ Cc, float* __restrict__ Ss) {
  const int l = blockIdx.x;
  const int j = threadIdx.x;
  const float* xr = x + (size_t)l * 1024;
  const float* wr = Wp + (size_t)j * 1024;
  float acc = bp[j];
  for (int k = 0; k < 1024; k += 4) {
    float4 xv = *(const float4*)(xr + k);
    float4 wv = *(const float4*)(wr + k);
    acc = fmaf(xv.x, wv.x, acc); acc = fmaf(xv.y, wv.y, acc);
    acc = fmaf(xv.z, wv.z, acc); acc = fmaf(xv.w, wv.w, acc);
  }
  float other = __shfl_xor(acc, 1, 32);
  if ((j & 1) == 0) {
    const int h = j >> 1;
    float c = acc, s = other;
    float nrm = fmaxf(sqrtf(c * c + s * s), 1e-6f);
    float g = 0.08f / (1.0f + __expf(-gamma[h]));
    float gs = sqrtf(g);
    Cc[(size_t)h * 2048 + l] = gs * c / nrm;
    Ss[(size_t)h * 2048 + l] = gs * s / nrm;
  }
}

// ---------------------------------------------------------------------------
// bf16 WMMA GEMM: C[2048][1024] = scale * (A[M][K] @ W[N][K]^T + bias)
// 256 threads / 8 waves; block tile 128x128; wave tile 32x64; double-buffered
// LDS K-loop; LDS-staged coalesced epilogue for bf16 modes.
// MODE 0: bf16 -> [H][L][64]; MODE 1: bf16 -> [H][64][L]; MODE 2: f32 [M][1024]
// ---------------------------------------------------------------------------
#define OTST 136   // epilogue staging stride (elements), 272B (16B multiple)

template <int MODE>
__global__ __launch_bounds__(256) void ia_gemm_kernel(
    const __bf16* __restrict__ A, const __bf16* __restrict__ W,
    const float* __restrict__ bias, void* __restrict__ outp, float scale) {
  const int K = 1024, NSTEP = K / 32;
  const int row0 = blockIdx.x * 128;
  const int col0 = blockIdx.y * 128;
  const int tid  = threadIdx.x;
  const int wave = tid >> 5, lane = tid & 31;
  const int lrow = lane & 15, lhal = lane >> 4;
  const int mrow0 = (wave >> 1) * 32;
  const int ncol0 = (wave & 1) * 64;

  __shared__ union {
    __bf16 tiles[2][2][128][48];   // [buf][A=0/B=1][row][elem]
    __bf16 ot[128][OTST];          // epilogue staging (aliases tiles)
  } sm;

  v8f acc[2][4];
#pragma unroll
  for (int mt = 0; mt < 2; ++mt)
#pragma unroll
    for (int nt = 0; nt < 4; ++nt) acc[mt][nt] = vzero8();

  const int r_ld = tid >> 1;
  const int c_ld = (tid & 1) * 16;
  uint4 ra0, ra1, rb0, rb1;

  auto gload = [&](int k0) {
    const uint4* sa = (const uint4*)(A + (size_t)(row0 + r_ld) * K + k0 + c_ld);
    ra0 = sa[0]; ra1 = sa[1];
    const uint4* sb = (const uint4*)(W + (size_t)(col0 + r_ld) * K + k0 + c_ld);
    rb0 = sb[0]; rb1 = sb[1];
  };
  auto sstore = [&](int buf) {
    *(uint4*)&sm.tiles[buf][0][r_ld][c_ld + 0] = ra0;
    *(uint4*)&sm.tiles[buf][0][r_ld][c_ld + 8] = ra1;
    *(uint4*)&sm.tiles[buf][1][r_ld][c_ld + 0] = rb0;
    *(uint4*)&sm.tiles[buf][1][r_ld][c_ld + 8] = rb1;
  };

  gload(0);
  sstore(0);

  for (int kt = 0; kt < NSTEP; ++kt) {
    __syncthreads();
    if (kt + 1 < NSTEP) gload((kt + 1) * 32);   // overlap next-tile loads
    const int cur = kt & 1;

    v16bf af[2];
#pragma unroll
    for (int mt = 0; mt < 2; ++mt) {
      const int arow = mrow0 + mt * 16 + lrow;
      const int ab   = lhal * 8;
      af[mt] = load_frag2(&sm.tiles[cur][0][arow][ab], &sm.tiles[cur][0][arow][ab + 16]);
    }
#pragma unroll
    for (int nt = 0; nt < 4; ++nt) {
      const int bcol = ncol0 + nt * 16 + lrow;
      v16bf bfv = load_bfrag(&sm.tiles[cur][1][bcol][lhal * 8]);
#pragma unroll
      for (int mt = 0; mt < 2; ++mt)
        acc[mt][nt] = __builtin_amdgcn_wmma_f32_16x16x32_bf16(
            false, af[mt], false, bfv, (short)0, acc[mt][nt], false, false);
    }
    if (kt + 1 < NSTEP) sstore((kt + 1) & 1);
  }

  if (MODE == 2) {
#pragma unroll
    for (int mt = 0; mt < 2; ++mt)
#pragma unroll
      for (int nt = 0; nt < 4; ++nt) {
        const int n = col0 + ncol0 + nt * 16 + lrow;
        const float bval = bias[n];
#pragma unroll
        for (int r = 0; r < 8; ++r) {
          const int m = row0 + mrow0 + mt * 16 + r + 8 * lhal;
          ((float*)outp)[(size_t)m * 1024 + n] = (acc[mt][nt][r] + bval) * scale;
        }
      }
    return;
  }

  // bf16 modes: stage block tile in LDS, then coalesced b128 global stores.
  __syncthreads();   // tiles no longer needed; ot aliases them
#pragma unroll
  for (int mt = 0; mt < 2; ++mt)
#pragma unroll
    for (int nt = 0; nt < 4; ++nt) {
      const int nloc = ncol0 + nt * 16 + lrow;
      const float bval = bias[col0 + nloc];
#pragma unroll
      for (int r = 0; r < 8; ++r) {
        const int mloc = mrow0 + mt * 16 + r + 8 * lhal;
        const __bf16 v = f2bf((acc[mt][nt][r] + bval) * scale);
        if (MODE == 0) sm.ot[mloc][nloc] = v;    // [m][n]
        else           sm.ot[nloc][mloc] = v;    // [n][m]
      }
    }
  __syncthreads();

  if (MODE == 0) {
    const int mloc = tid >> 1;
    const int nseg = (tid & 1) * 64;
    const int h = (col0 + nseg) >> 6;
    __bf16* dst = (__bf16*)outp + ((size_t)h * 2048 + row0 + mloc) * 64;
#pragma unroll
    for (int i = 0; i < 8; ++i)
      ((uint4*)dst)[i] = *(const uint4*)&sm.ot[mloc][nseg + i * 8];
  } else {
    const int nloc = tid >> 1;
    const int mseg = (tid & 1) * 64;
    const int n = col0 + nloc;
    const int h = n >> 6, d = n & 63;
    __bf16* dst = (__bf16*)outp + ((size_t)h * 64 + d) * 2048 + row0 + mseg;
#pragma unroll
    for (int i = 0; i < 8; ++i)
      ((uint4*)dst)[i] = *(const uint4*)&sm.ot[nloc][mseg + i * 8];
  }
}

// ---------------------------------------------------------------------------
// Flash attention with rank-2 interference bias.
// Q is pre-scaled by 1/sqrt(HD), so S = Q'K^T + bias comes straight from the
// WMMA chain with the rank-2 bias supplied as the C operand.
// Grid (L/128, H); 256 threads / 8 waves; wave = 16 query rows.
// K / V^T tiles double-buffered via Tensor Data Mover (async DMA overlap).
// ---------------------------------------------------------------------------
#define KST 72

__global__ __launch_bounds__(256) void ia_attn_kernel(
    const __bf16* __restrict__ Q, const __bf16* __restrict__ Kh,
    const __bf16* __restrict__ Vt, const float* __restrict__ Cc,
    const float* __restrict__ Ss, __bf16* __restrict__ Ao) {
  const int L = 2048, NKB = L / 64;
  const int h  = blockIdx.y;
  const int q0 = blockIdx.x * 128;
  const int tid = threadIdx.x;
  const int wave = tid >> 5, lane = tid & 31;
  const int lrow = lane & 15, lhal = lane >> 4;

  __shared__ __bf16 Ks[2][64][KST];
  __shared__ __bf16 Vs[2][64][KST];
  __shared__ __bf16 Ps[128][KST];
  __shared__ float  cks[2][64], sks[2][64];

  const __bf16* Qh  = Q  + (size_t)h * L * 64;
  const __bf16* Khh = Kh + (size_t)h * L * 64;
  const __bf16* Vth = Vt + (size_t)h * 64 * L;

  // Q fragments (loop invariant)
  v16bf qf[2];
  {
    const int qrow = q0 + wave * 16 + lrow;
#pragma unroll
    for (int kd = 0; kd < 2; ++kd) {
      const int base = kd * 32 + lhal * 8;
      qf[kd] = load_frag2(Qh + (size_t)qrow * 64 + base,
                          Qh + (size_t)qrow * 64 + base + 16);
    }
  }

  float cq[8], sq[8], m[8], lsum[8];
#pragma unroll
  for (int r = 0; r < 8; ++r) {
    const int qr = q0 + wave * 16 + r + 8 * lhal;
    cq[r] = Cc[(size_t)h * L + qr];
    sq[r] = Ss[(size_t)h * L + qr];
    m[r] = -1e30f; lsum[r] = 0.0f;
  }
  v8f Oacc[4];
#pragma unroll
  for (int t = 0; t < 4; ++t) Oacc[t] = vzero8();

  auto issue_stage = [&](int kb, int buf) {
    const int k0 = kb * 64;
    if (wave == 0) {
      tdm_load_tile_2d((unsigned)(size_t)(void*)&Ks[buf][0][0],
                       (unsigned long long)(size_t)(Khh + (size_t)k0 * 64),
                       64, (unsigned)L, 64, 64, 64ull);
    } else if (wave == 1) {
      tdm_load_tile_2d((unsigned)(size_t)(void*)&Vs[buf][0][0],
                       (unsigned long long)(size_t)(Vth + (size_t)k0),
                       (unsigned)L, 64, 64, 64, (unsigned long long)L);
    }
    if (tid < 64) {
      cks[buf][tid] = Cc[(size_t)h * L + k0 + tid];
      sks[buf][tid] = Ss[(size_t)h * L + k0 + tid];
    }
  };

  issue_stage(0, 0);                   // prologue DMA into buffer 0

  for (int kb = 0; kb < NKB; ++kb) {
    if (wave < 2) __builtin_amdgcn_s_wait_tensorcnt(0);
    __syncthreads();                   // current buffer's DMA visible to all
    if (kb + 1 < NKB) issue_stage(kb + 1, (kb + 1) & 1);   // overlap next DMA
    const int cur = kb & 1;

    // S = Q'K^T + (C_i C_j + S_i S_j): rank-2 bias as the WMMA C operand
    v8f st[4];
#pragma unroll
    for (int nt = 0; nt < 4; ++nt) {
      const int bcol = nt * 16 + lrow;
      const float ck = cks[cur][bcol], sk = sks[cur][bcol];
      v8f s;
#pragma unroll
      for (int r = 0; r < 8; ++r)
        s[r] = fmaf(cq[r], ck, sq[r] * sk);
#pragma unroll
      for (int kd = 0; kd < 2; ++kd) {
        v16bf bfv = load_bfrag(&Ks[cur][bcol][kd * 32 + lhal * 8]);
        s = __builtin_amdgcn_wmma_f32_16x16x32_bf16(
            false, qf[kd], false, bfv, (short)0, s, false, false);
      }
      st[nt] = s;
    }

    // Online softmax
    float mn[8], sc[8];
#pragma unroll
    for (int r = 0; r < 8; ++r) {
      float rm = fmaxf(fmaxf(st[0][r], st[1][r]), fmaxf(st[2][r], st[3][r]));
      rm = fmaxf(rm, __shfl_xor(rm, 1, 32));
      rm = fmaxf(rm, __shfl_xor(rm, 2, 32));
      rm = fmaxf(rm, __shfl_xor(rm, 4, 32));
      rm = fmaxf(rm, __shfl_xor(rm, 8, 32));
      mn[r] = fmaxf(m[r], rm);
      sc[r] = __expf(m[r] - mn[r]);
      m[r]  = mn[r];
    }
    float rs[8];
#pragma unroll
    for (int r = 0; r < 8; ++r) rs[r] = 0.0f;
#pragma unroll
    for (int nt = 0; nt < 4; ++nt) {
      const int prow = wave * 16 + 8 * lhal;
      const int pcol = nt * 16 + lrow;
#pragma unroll
      for (int r = 0; r < 8; ++r) {
        float pv = __expf(st[nt][r] - mn[r]);
        rs[r] += pv;
        Ps[prow + r][pcol] = f2bf(pv);
      }
    }
#pragma unroll
    for (int r = 0; r < 8; ++r) {
      rs[r] += __shfl_xor(rs[r], 1, 32);
      rs[r] += __shfl_xor(rs[r], 2, 32);
      rs[r] += __shfl_xor(rs[r], 4, 32);
      rs[r] += __shfl_xor(rs[r], 8, 32);
      lsum[r] = lsum[r] * sc[r] + rs[r];
    }
#pragma unroll
    for (int t = 0; t < 4; ++t)
#pragma unroll
      for (int r = 0; r < 8; ++r) Oacc[t][r] *= sc[r];

    // O += P @ V
#pragma unroll
    for (int jb = 0; jb < 2; ++jb) {
      const int prow  = wave * 16 + lrow;
      const int pbase = jb * 32 + lhal * 8;
      v16bf pf = load_frag2(&Ps[prow][pbase], &Ps[prow][pbase + 16]);
#pragma unroll
      for (int dt = 0; dt < 4; ++dt) {
        const int dcol = dt * 16 + lrow;
        v16bf vf = load_bfrag(&Vs[cur][dcol][jb * 32 + lhal * 8]);
        Oacc[dt] = __builtin_amdgcn_wmma_f32_16x16x32_bf16(
            false, pf, false, vf, (short)0, Oacc[dt], false, false);
      }
    }
  }

  // Normalize and store bf16 to [L][D]
#pragma unroll
  for (int r = 0; r < 8; ++r) {
    const float inv = 1.0f / lsum[r];
    const int qr = q0 + wave * 16 + r + 8 * lhal;
#pragma unroll
    for (int dt = 0; dt < 4; ++dt) {
      const int d = dt * 16 + lrow;
      Ao[(size_t)qr * 1024 + h * 64 + d] = f2bf(Oacc[dt][r] * inv);
    }
  }
}

// ---------------------------------------------------------------------------
extern "C" void kernel_launch(void* const* d_in, const int* in_sizes, int n_in,
                              void* d_out, int out_size, void* d_ws, size_t ws_size,
                              hipStream_t stream) {
  const float* x     = (const float*)d_in[0];
  const float* Wq    = (const float*)d_in[1];
  const float* bq    = (const float*)d_in[2];
  const float* Wk    = (const float*)d_in[3];
  const float* bk    = (const float*)d_in[4];
  const float* Wv    = (const float*)d_in[5];
  const float* bv    = (const float*)d_in[6];
  const float* Wo    = (const float*)d_in[7];
  const float* bo    = (const float*)d_in[8];
  const float* Wp    = (const float*)d_in[9];
  const float* bp    = (const float*)d_in[10];
  const float* gamma = (const float*)d_in[11];
  float* out = (float*)d_out;

  char* ws = (char*)d_ws;
  __bf16* xb  = (__bf16*)(ws);
  __bf16* Wqb = (__bf16*)(ws + ( 4u << 20));
  __bf16* Wkb = (__bf16*)(ws + ( 6u << 20));
  __bf16* Wvb = (__bf16*)(ws + ( 8u << 20));
  __bf16* Wob = (__bf16*)(ws + (10u << 20));
  __bf16* qh  = (__bf16*)(ws + (12u << 20));   // [H][L][64], pre-scaled 1/8
  __bf16* kh  = (__bf16*)(ws + (16u << 20));   // [H][L][64]
  __bf16* vt  = (__bf16*)(ws + (20u << 20));   // [H][64][L]
  __bf16* ao  = (__bf16*)(ws + (24u << 20));   // [L][D]
  float*  Cc  = (float*) (ws + (28u << 20));
  float*  Ss  = (float*) (ws + (28u << 20) + (1u << 18));

  ia_cvt_kernel<<<1024, 256, 0, stream>>>((const float4*)x,  (uint2*)xb,  (2048 * 1024) / 4);
  ia_cvt_kernel<<<512,  256, 0, stream>>>((const float4*)Wq, (uint2*)Wqb, (1024 * 1024) / 4);
  ia_cvt_kernel<<<512,  256, 0, stream>>>((const float4*)Wk, (uint2*)Wkb, (1024 * 1024) / 4);
  ia_cvt_kernel<<<512,  256, 0, stream>>>((const float4*)Wv, (uint2*)Wvb, (1024 * 1024) / 4);
  ia_cvt_kernel<<<512,  256, 0, stream>>>((const float4*)Wo, (uint2*)Wob, (1024 * 1024) / 4);

  ia_phase_kernel<<<2048, 32, 0, stream>>>(x, Wp, bp, gamma, Cc, Ss);

  dim3 gg(2048 / 128, 1024 / 128);
  ia_gemm_kernel<0><<<gg, 256, 0, stream>>>(xb, Wqb, bq, (void*)qh, 0.125f);  // Q pre-scaled
  ia_gemm_kernel<0><<<gg, 256, 0, stream>>>(xb, Wkb, bk, (void*)kh, 1.0f);
  ia_gemm_kernel<1><<<gg, 256, 0, stream>>>(xb, Wvb, bv, (void*)vt, 1.0f);

  ia_attn_kernel<<<dim3(2048 / 128, 16), 256, 0, stream>>>(qh, kh, vt, Cc, Ss, ao);

  ia_gemm_kernel<2><<<gg, 256, 0, stream>>>(ao, Wob, bo, (void*)out, 1.0f);
}
